// YoloNASOBBLoss_42073499631800
// MI455X (gfx1250) — matrix operand order repaired and checked by reading, not compiled, not run on hardware
//
#include <hip/hip_runtime.h>
#include <hip/hip_bf16.h>

typedef __attribute__((ext_vector_type(16))) _Float16 v16h;
typedef __attribute__((ext_vector_type(8)))  float    v8f;

#define NB   8
#define NL   21504
#define NC   15
#define NG   60
#define NTOP 13
#define BL   (NB * NL)

// gt descriptor layout (12 floats):
// 0 mnx, 1 mxx, 2 mny, 3 mxy, 4 area, 5 cx, 6 cy, 7 cos, 8 sin, 9 hw, 10 hh, 11 pad

__device__ __forceinline__ void rbox_aabb(float cx, float cy, float w, float h, float r,
                                          float& mnx, float& mxx, float& mny, float& mxy) {
  float c = cosf(r), s = sinf(r);
  float dx = 0.5f * w * c, dy = 0.5f * h * s;       // matches reference _box_min_max
  // corner offsets (ox,oy) in {(-dx,-dy),(dx,-dy),(dx,dy),(-dx,dy)} rotated by r
  float x1 = -dx * c + dy * s, y1 = -dx * s - dy * c;
  float x2 =  dx * c + dy * s, y2 =  dx * s - dy * c;
  float x3 =  dx * c - dy * s, y3 =  dx * s + dy * c;
  float x4 = -dx * c - dy * s, y4 = -dx * s + dy * c;
  mnx = cx + fminf(fminf(x1, x2), fminf(x3, x4));
  mxx = cx + fmaxf(fmaxf(x1, x2), fmaxf(x3, x4));
  mny = cy + fminf(fminf(y1, y2), fminf(y3, y4));
  mxy = cy + fmaxf(fmaxf(y1, y2), fmaxf(y3, y4));
}

__device__ __forceinline__ float iou_calc(const float* d, float pmn, float pmx,
                                          float pny, float pxy, float pa) {
  float iw = fmaxf(fminf(d[1], pmx) - fmaxf(d[0], pmn), 0.0f);
  float ih = fmaxf(fminf(d[3], pxy) - fmaxf(d[2], pny), 0.0f);
  float inter = iw * ih;
  float u = d[4] + pa - inter + 1e-9f;
  return fminf(fmaxf(inter / u, 0.0f), 1.0f);
}

__global__ void k_zero(unsigned* __restrict__ counts, float* __restrict__ maxes) {
  int i = blockIdx.x * 256 + threadIdx.x;
  if (i < BL) counts[i] = 0u;
  if (i < 2 * NB * NG) maxes[i] = 0.0f;   // maxm (480) ++ maxiou (480) contiguous
}

__global__ void k_desc(const float* __restrict__ gtb, const float* __restrict__ padm,
                       float* __restrict__ desc) {
  int i = blockIdx.x * 256 + threadIdx.x;
  if (i >= NB * NG) return;
  float cx = gtb[i * 5 + 0], cy = gtb[i * 5 + 1];
  float w  = gtb[i * 5 + 2], h  = gtb[i * 5 + 3], r = gtb[i * 5 + 4];
  float mnx, mxx, mny, mxy;
  rbox_aabb(cx, cy, w, h, r, mnx, mxx, mny, mxy);
  float* d = desc + (size_t)i * 12;
  d[0] = mnx; d[1] = mxx; d[2] = mny; d[3] = mxy;
  d[4] = w * h; d[5] = cx; d[6] = cy;
  d[7] = cosf(r); d[8] = sinf(r);
  d[9] = 0.5f * w; d[10] = 0.5f * h;
  d[11] = padm[i];
}

__global__ void k_pred(const float* __restrict__ prb,
                       float* __restrict__ pmnx, float* __restrict__ pmxx,
                       float* __restrict__ pmny, float* __restrict__ pmxy,
                       float* __restrict__ parea) {
  size_t i = (size_t)blockIdx.x * 256 + threadIdx.x;
  if (i >= BL) return;
  float cx = prb[i * 5 + 0], cy = prb[i * 5 + 1];
  float w  = prb[i * 5 + 2], h  = prb[i * 5 + 3], r = prb[i * 5 + 4];
  float mnx, mxx, mny, mxy;
  rbox_aabb(cx, cy, w, h, r, mnx, mxx, mny, mxy);
  pmnx[i] = mnx; pmxx[i] = mxx; pmny[i] = mny; pmxy[i] = mxy;
  parea[i] = w * h;
}

// Fused: WMMA one-hot class gather + AABB IoU + point-in-rbox -> metric (B,NG,NL)
__launch_bounds__(256)
__global__ void k_main(const float* __restrict__ ps, const float* __restrict__ ap,
                       const int* __restrict__ glab, const float* __restrict__ desc,
                       const float* __restrict__ pmnx, const float* __restrict__ pmxx,
                       const float* __restrict__ pmny, const float* __restrict__ pmxy,
                       const float* __restrict__ parea, float* __restrict__ metric) {
  __shared__ float sd[NG * 12];
  __shared__ int   sl[NG];
  int tid = threadIdx.x;
  int b = blockIdx.y;
  for (int t = tid; t < NG * 12; t += 256) sd[t] = desc[(size_t)b * NG * 12 + t];
  if (tid < NG) sl[tid] = glab[b * NG + tid];
  __syncthreads();

  int lane = tid & 31;
  int wave = tid >> 5;
  int n = lane & 15;
  int l = blockIdx.x * 128 + wave * 16 + n;
  size_t bl = (size_t)b * NL + l;

  // B operand (32x16 f16): lanes 0-15 hold column n's classes K=0..15 (2/VGPR);
  // lanes 16-31 (K=16..31) are zero (C=15 padded to K=32).
  v16h bb = {};
  if (lane < 16) {
    const float* sp = ps + bl * NC;
    #pragma unroll
    for (int k = 0; k < NC; ++k) bb[k] = (_Float16)sp[k];
  }

  float pmn = pmnx[bl], pmx = pmxx[bl], pny = pmny[bl], pxy = pmxy[bl], pa = parea[bl];
  float px = ap[2 * l], py = ap[2 * l + 1];

  int kbase = (lane < 16) ? 0 : 16;  // A lane-half K offset
  int ghalf = (lane < 16) ? 0 : 8;   // D lane-half row offset

  for (int gt = 0; gt < 4; ++gt) {
    // A operand (16x32 f16): row M=lane&15, elems 0..15 cover K=kbase..kbase+15
    int M = lane & 15;
    int g = gt * 16 + M;
    int lbl = (g < NG) ? sl[g] : -1;
    v16h aa = {};
    #pragma unroll
    for (int i = 0; i < 16; ++i) aa[i] = (_Float16)((kbase + i == lbl) ? 1.0f : 0.0f);

    v8f acc = {};
    acc = __builtin_amdgcn_wmma_f32_16x16x32_f16(false, aa, false, bb,
                                                 (short)0, acc, false, false);
    // D layout: VGPR r, lanes0-15 -> row gt*16+r, lanes16-31 -> row gt*16+r+8
    #pragma unroll
    for (int r = 0; r < 8; ++r) {
      int gg = gt * 16 + r + ghalf;
      if (gg >= NG) continue;
      const float* d = &sd[gg * 12];
      float iou = iou_calc(d, pmn, pmx, pny, pxy, pa);
      float dxp = px - d[5], dyp = py - d[6];
      float xl =  dxp * d[7] + dyp * d[8];
      float yl = -dxp * d[8] + dyp * d[7];
      bool inside = (fabsf(xl) <= d[9]) && (fabsf(yl) <= d[10]);
      float i2 = iou * iou;
      float m = acc[r] * i2 * i2 * i2;              // cls^1 * iou^6
      metric[((size_t)b * NG + gg) * NL + l] = inside ? m : 0.0f;
    }
  }
}

// One block per (b,g) row: 13 exact argmax rounds (ties -> lowest index),
// winners inside the gt box bump per-anchor positive counts.
__global__ void k_topk(const float* __restrict__ metric, const float* __restrict__ desc,
                       const float* __restrict__ ap, unsigned* __restrict__ counts,
                       int* __restrict__ gidx) {
  int row = blockIdx.x;                 // b*NG + g
  int b = row / NG, g = row % NG;
  __shared__ float sv[256];
  __shared__ int   si[256];
  __shared__ int   chosen[NTOP];
  __shared__ float gd[12];
  int tid = threadIdx.x;
  if (tid < 12) gd[tid] = desc[(size_t)row * 12 + tid];
  __syncthreads();
  if (gd[11] == 0.0f) return;           // pad_gt_mask == 0 -> contributes nothing

  const float* mrow = metric + (size_t)row * NL;
  for (int t = 0; t < NTOP; ++t) {
    float bv = -1.0f; int bi = 0x40000000;
    for (int l = tid; l < NL; l += 256) {
      float v = mrow[l];
      bool skip = false;
      for (int j = 0; j < t; ++j) skip |= (chosen[j] == l);
      if (skip) continue;
      if (v > bv || (v == bv && l < bi)) { bv = v; bi = l; }
    }
    sv[tid] = bv; si[tid] = bi;
    __syncthreads();
    for (int s = 128; s > 0; s >>= 1) {
      if (tid < s) {
        float ov = sv[tid + s]; int oi = si[tid + s];
        if (ov > sv[tid] || (ov == sv[tid] && oi < si[tid])) { sv[tid] = ov; si[tid] = oi; }
      }
      __syncthreads();
    }
    if (tid == 0) chosen[t] = si[0];
    __syncthreads();
  }
  if (tid < NTOP) {
    int l = chosen[tid];
    float px = ap[2 * l], py = ap[2 * l + 1];
    float dxp = px - gd[5], dyp = py - gd[6];
    float xl =  dxp * gd[7] + dyp * gd[8];
    float yl = -dxp * gd[8] + dyp * gd[7];
    if (fabsf(xl) <= gd[9] && fabsf(yl) <= gd[10]) {
      atomicAdd(&counts[(size_t)b * NL + l], 1u);
      gidx[(size_t)b * NL + l] = g;     // only consumed when count==1 (single writer)
    }
  }
}

// Per anchor: resolve assigned gt; accumulate per-gt max metric / max IoU.
__global__ void k_assign(const unsigned* __restrict__ counts, const int* __restrict__ gidx,
                         const float* __restrict__ desc, const int* __restrict__ glab,
                         const float* __restrict__ ps,
                         const float* __restrict__ pmnx, const float* __restrict__ pmxx,
                         const float* __restrict__ pmny, const float* __restrict__ pmxy,
                         const float* __restrict__ parea,
                         int* __restrict__ ag, float* __restrict__ m_arr,
                         unsigned* __restrict__ maxm, unsigned* __restrict__ maxiou) {
  __shared__ float sd[NG * 12];
  __shared__ int   sl[NG];
  int tid = threadIdx.x;
  int b = (int)(((size_t)blockIdx.x * 256) / NL);   // uniform per block (NL % 256 == 0)
  for (int t = tid; t < NG * 12; t += 256) sd[t] = desc[(size_t)b * NG * 12 + t];
  if (tid < NG) sl[tid] = glab[b * NG + tid];
  __syncthreads();

  size_t i = (size_t)blockIdx.x * 256 + tid;
  int l = (int)(i % NL);
  unsigned c = counts[i];
  if (c == 0u) { ag[i] = -1; m_arr[i] = 0.0f; return; }

  float pmn = pmnx[i], pmx = pmxx[i], pny = pmny[i], pxy = pmxy[i], pa = parea[i];
  int gstar; float ioustar;
  if (c == 1u) {
    gstar = gidx[i];
    ioustar = iou_calc(&sd[gstar * 12], pmn, pmx, pny, pxy, pa);
  } else {
    float bv = -1.0f; int bi = 0;
    for (int g = 0; g < NG; ++g) {
      float v = iou_calc(&sd[g * 12], pmn, pmx, pny, pxy, pa);
      if (v > bv) { bv = v; bi = g; }   // first-max tie rule == jnp.argmax
    }
    gstar = bi; ioustar = bv;
  }
  float cls = ps[((size_t)b * NL + l) * NC + sl[gstar]];
  float i2 = ioustar * ioustar;
  float m = cls * i2 * i2 * i2;
  ag[i] = gstar;
  m_arr[i] = m;
  // values are >= 0 -> uint ordering matches float ordering
  atomicMax(&maxm[b * NG + gstar], __float_as_uint(m));
  atomicMax(&maxiou[b * NG + gstar], __float_as_uint(ioustar));
}

__global__ void k_final(const int* __restrict__ ag, const float* __restrict__ m_arr,
                        const float* __restrict__ maxm, const float* __restrict__ maxiou,
                        const int* __restrict__ glab, const float* __restrict__ gtb,
                        const int* __restrict__ gcrowd, const int* __restrict__ bgp,
                        float* __restrict__ out) {
  size_t i = (size_t)blockIdx.x * 256 + threadIdx.x;
  if (i >= BL) return;
  int b = (int)(i / NL);
  int bgi = bgp[0];
  int g = ag[i];
  float per = 0.0f;
  int lbl = bgi, gout = 0;
  if (g >= 0) {
    per = m_arr[i] / (maxm[b * NG + g] + 1e-9f) * maxiou[b * NG + g];
    lbl = glab[b * NG + g];
    gout = g;
  }
  int crowd = gcrowd[b * NG + gout];

  float* o_lab = out;
  float* o_rb  = out + (size_t)BL;
  float* o_sc  = out + (size_t)BL * 6;
  float* o_gi  = out + (size_t)BL * 21;
  float* o_cr  = out + (size_t)BL * 22;

  o_lab[i] = (float)lbl;
  const float* rb = gtb + (size_t)(b * NG + gout) * 5;
  #pragma unroll
  for (int j = 0; j < 5; ++j) o_rb[i * 5 + j] = rb[j];
  float sval = (crowd == 0) ? per : 0.0f;
  #pragma unroll
  for (int k = 0; k < NC; ++k)
    o_sc[i * NC + k] = (g >= 0 && k == lbl) ? sval : 0.0f;
  o_gi[i] = (float)gout;
  o_cr[i] = (float)crowd;
}

extern "C" void kernel_launch(void* const* d_in, const int* in_sizes, int n_in,
                              void* d_out, int out_size, void* d_ws, size_t ws_size,
                              hipStream_t stream) {
  (void)in_sizes; (void)n_in; (void)out_size; (void)ws_size;
  const float* ps     = (const float*)d_in[0];   // (B,L,C)
  const float* prb    = (const float*)d_in[1];   // (B,L,5)
  const float* ap     = (const float*)d_in[2];   // (L,2)
  const int*   glab   = (const int*)  d_in[3];   // (B,n,1)
  const float* gtb    = (const float*)d_in[4];   // (B,n,5)
  const int*   gcrowd = (const int*)  d_in[6];   // (B,n,1)
  const float* padm   = (const float*)d_in[7];   // (B,n,1)
  const int*   bgp    = (const int*)  d_in[8];   // scalar
  float* out = (float*)d_out;

  float* wsf = (float*)d_ws;
  float* pmnx  = wsf + 0 * (size_t)BL;
  float* pmxx  = wsf + 1 * (size_t)BL;
  float* pmny  = wsf + 2 * (size_t)BL;
  float* pmxy  = wsf + 3 * (size_t)BL;
  float* parea = wsf + 4 * (size_t)BL;
  float* metric = wsf + 5 * (size_t)BL;            // NG*BL floats (~41 MB, L2-resident)
  float* desc   = metric + (size_t)NG * BL;        // NB*NG*12
  unsigned* counts = (unsigned*)(desc + NB * NG * 12);
  int* gidx  = (int*)(counts + BL);
  int* ag    = gidx + BL;
  float* m_arr  = (float*)(ag + BL);
  float* maxm   = m_arr + BL;                      // NB*NG
  float* maxiou = maxm + NB * NG;                  // NB*NG (contiguous after maxm)

  const int blk = 256;
  const int nBL = (BL + blk - 1) / blk;            // 672

  k_zero<<<nBL, blk, 0, stream>>>(counts, maxm);
  k_desc<<<(NB * NG + blk - 1) / blk, blk, 0, stream>>>(gtb, padm, desc);
  k_pred<<<nBL, blk, 0, stream>>>(prb, pmnx, pmxx, pmny, pmxy, parea);
  k_main<<<dim3(NL / 128, NB), blk, 0, stream>>>(ps, ap, glab, desc,
                                                 pmnx, pmxx, pmny, pmxy, parea, metric);
  k_topk<<<NB * NG, blk, 0, stream>>>(metric, desc, ap, counts, gidx);
  k_assign<<<nBL, blk, 0, stream>>>(counts, gidx, desc, glab, ps,
                                    pmnx, pmxx, pmny, pmxy, parea,
                                    ag, m_arr, (unsigned*)maxm, (unsigned*)maxiou);
  k_final<<<nBL, blk, 0, stream>>>(ag, m_arr, maxm, maxiou, glab, gtb, gcrowd, bgp, out);
}
